// Encoder_64149631533576
// MI455X (gfx1250) — matrix operand order repaired and checked by reading, not compiled, not run on hardware
//
#include <hip/hip_runtime.h>
#include <cstdint>

typedef __attribute__((ext_vector_type(16))) __bf16 v16bf;
typedef __attribute__((ext_vector_type(8)))  __bf16 v8bf;
typedef __attribute__((ext_vector_type(8)))  float  v8f;
typedef __attribute__((ext_vector_type(4)))  int    v4i;

// pointee-address-space-qualified pointers for the async-LDS builtin
typedef __attribute__((address_space(1))) v4i* gv4i_p;   // global int4*
typedef __attribute__((address_space(3))) v4i* lv4i_p;   // LDS int4*

#define HDIM   512
#define BSZ    128
#define TSTEPS 512
#define INDIM  128

#if __has_builtin(__builtin_amdgcn_global_load_async_to_lds_b128)
#define USE_ASYNC_LDS 1
__device__ __forceinline__ void async_copy16(const void* gsrc, void* ldst) {
    __builtin_amdgcn_global_load_async_to_lds_b128(
        (gv4i_p)(uintptr_t)gsrc, (lv4i_p)(uintptr_t)ldst, 0, 0);
}
#else
#define USE_ASYNC_LDS 0
#endif

// ---------------------------------------------------------------------------
// small prep kernels
// ---------------------------------------------------------------------------
__global__ void zero_f32_kernel(float* __restrict__ p, int n) {
    int i = blockIdx.x * blockDim.x + threadIdx.x;
    if (i < n) p[i] = 0.0f;
}

__global__ void zero_bf16_kernel(__bf16* __restrict__ p, int n) {
    int i = blockIdx.x * blockDim.x + threadIdx.x;
    if (i < n) p[i] = (__bf16)0.0f;
}

__global__ void f32_to_bf16_kernel(const float* __restrict__ src,
                                   __bf16* __restrict__ dst, long n) {
    long i = (long)blockIdx.x * blockDim.x + threadIdx.x;
    if (i < n) dst[i] = (__bf16)src[i];
}

__global__ void bf16_to_f32_kernel(const __bf16* __restrict__ src,
                                   float* __restrict__ dst, int n) {
    int i = blockIdx.x * blockDim.x + threadIdx.x;
    if (i < n) dst[i] = (float)src[i];
}

// Wcat[g][k] = k < KIN ? W_ih[g][k] : W_hh[g][k-KIN], converted to bf16.
__global__ void prep_weights_kernel(const float* __restrict__ Wih,
                                    const float* __restrict__ Whh,
                                    __bf16* __restrict__ Wcat,
                                    int K, int KIN, long total) {
    long i = (long)blockIdx.x * blockDim.x + threadIdx.x;
    if (i >= total) return;
    int row = (int)(i / K);
    int col = (int)(i - (long)row * K);
    float v = (col < KIN) ? Wih[(long)row * KIN + col]
                          : Whh[(long)row * HDIM + (col - KIN)];
    Wcat[i] = (__bf16)v;
}

__global__ void prep_bias_kernel(const float* __restrict__ bih,
                                 const float* __restrict__ bhh,
                                 float* __restrict__ bias, int n) {
    int i = blockIdx.x * blockDim.x + threadIdx.x;
    if (i < n) bias[i] = bih[i] + bhh[i];
}

// ---------------------------------------------------------------------------
// One LSTM timestep for one layer.
//   gates = [in_t | h_in] (B x K, bf16, async-staged in LDS) @ Wcat^T + bias
//
// Grid = 128 WGs x 256 threads (8 waves). WG bx: ib = bx>>4 (batch tile),
// jp = bx&15. Waves are split into 2 tile-groups (tg = wave>>2) covering
// jb = jp*2+tg, and 4 k-quarter waves (kw = wave&3) that each accumulate all
// 4 gate tiles (i,f,g,o) over K/4. Partials are combined deterministically
// through LDS, then the 128 lanes of each group apply the cell update.
// h state is kept in bf16 (it only ever feeds the bf16 WMMA, so this loses
// no precision vs converting at staging time); c state stays fp32.
// ---------------------------------------------------------------------------
template <int K, int KIN>
__global__ __launch_bounds__(256) void lstm_step_kernel(
    const __bf16* __restrict__ in, long in_row_stride, long in_t_off,
    const __bf16* __restrict__ h_in, __bf16* __restrict__ h_out,
    float* __restrict__ c,
    const __bf16* __restrict__ Wcat, const float* __restrict__ bias)
{
    constexpr int KQ     = K / 4;     // k-range per wave
    constexpr int KSTEPS = KQ / 32;   // WMMA k-steps per wave
    constexpr int UROW   = K / 8;     // 16-byte units per A row

    __shared__ __attribute__((aligned(16))) __bf16 Alds[16 * K];
    __shared__ float Sred[2][4][4][256];   // [group][gate][kquarter][m*16+n]

    const int tid  = threadIdx.x;
    const int wave = tid >> 5;
    const int lane = tid & 31;
    const int half = lane >> 4;   // 16-lane group within wave
    const int r    = lane & 15;   // A-row / B-column within tile
    const int tg   = wave >> 2;   // tile-group within WG (0..1)
    const int kw   = wave & 3;    // k-quarter (0..3)

    const int ib = blockIdx.x >> 4;        // 0..7  batch tile
    const int jp = blockIdx.x & 15;        // 0..15
    const int jb = jp * 2 + tg;            // 0..31 h-column tile
    const int n  = jb * 16 + r;            // this lane's B column

    // B fragment base pointers (row n of W == column n of B), k-quarter offset
    const __bf16* w[4];
    #pragma unroll
    for (int g = 0; g < 4; ++g)
        w[g] = Wcat + (size_t)(g * HDIM + n) * K + kw * KQ;

    // Prefetch first B fragments (independent of LDS staging; overlaps it).
    union bfrag { v16bf v; v8bf h[2]; };
    const int kb0 = half * 16;   // B: lane holds col n, contiguous K=kb0..kb0+15
    bfrag cur[4], nxt[4];
    #pragma unroll
    for (int g = 0; g < 4; ++g) {
        cur[g].h[0] = *(const v8bf*)(w[g] + kb0);
        cur[g].h[1] = *(const v8bf*)(w[g] + kb0 + 8);
        nxt[g] = cur[g];
    }

    // --- stage A = [x_t | h_prev] (already bf16) rows into LDS -------------
    // Pure 16-byte copies; on gfx1250 use async global->LDS (no VGPR bounce).
    for (int u = tid; u < 16 * UROW; u += 256) {
        const int row = u / UROW;
        const int col = (u - row * UROW) * 8;
        const int b   = ib * 16 + row;
        const __bf16* src = (col < KIN)
            ? in + (long)b * in_row_stride + in_t_off + col
            : h_in + (long)b * HDIM + (col - KIN);
        __bf16* dst = Alds + row * K + col;
#if USE_ASYNC_LDS
        async_copy16(src, dst);
#else
        *(v8bf*)dst = *(const v8bf*)src;
#endif
    }
#if USE_ASYNC_LDS
#if __has_builtin(__builtin_amdgcn_s_wait_asynccnt)
    __builtin_amdgcn_s_wait_asynccnt(0);
#else
    asm volatile("s_wait_asynccnt 0x0" ::: "memory");
#endif
#endif
    __syncthreads();

    // --- K/4 GEMM with software-pipelined (double-buffered) B fragments ---
    v8f acc[4] = {v8f{}, v8f{}, v8f{}, v8f{}};
    const __bf16* arow = Alds + r * K + kw * KQ;

    #pragma unroll
    for (int s = 0; s < KSTEPS; ++s) {
        const int k0 = s * 32;
        if (s + 1 < KSTEPS) {
            const int kn = k0 + 32 + kb0;
            #pragma unroll
            for (int g = 0; g < 4; ++g) {
                nxt[g].h[0] = *(const v8bf*)(w[g] + kn);
                nxt[g].h[1] = *(const v8bf*)(w[g] + kn + 8);
            }
        }
        // A 16x32 bf16: lane(half,r) holds row r, contiguous K = k0+16h..k0+16h+15
        const int ka = k0 + half * 16;
        union { v16bf v; v8bf h[2]; } a;
        a.h[0] = *(const v8bf*)(arow + ka);
        a.h[1] = *(const v8bf*)(arow + ka + 8);

        #pragma unroll
        for (int g = 0; g < 4; ++g)
            acc[g] = __builtin_amdgcn_wmma_f32_16x16x32_bf16(
                false, a.v, false, cur[g].v, (short)0, acc[g], false, false);
        #pragma unroll
        for (int g = 0; g < 4; ++g) cur[g] = nxt[g];
    }

    // --- deterministic cross-wave reduction via LDS ------------------------
    // C/D layout: VGPR rr holds (m = rr + 8*half, n = r)
    #pragma unroll
    for (int g = 0; g < 4; ++g) {
        #pragma unroll
        for (int rr = 0; rr < 8; ++rr)
            Sred[tg][g][kw][(rr + 8 * half) * 16 + r] = acc[g][rr];
    }
    __syncthreads();

    // --- cell update: 128 lanes per group, 2 elements each -----------------
    #pragma unroll
    for (int ee = 0; ee < 2; ++ee) {
        const int e  = (kw * 32 + lane) * 2 + ee;  // 0..255 within tile
        const int m  = e >> 4;
        const int ng = jb * 16 + (e & 15);
        float sI = 0.f, sF = 0.f, sG = 0.f, sO = 0.f;
        #pragma unroll
        for (int q = 0; q < 4; ++q) {   // fixed-order sum -> deterministic
            sI += Sred[tg][0][q][e];
            sF += Sred[tg][1][q][e];
            sG += Sred[tg][2][q][e];
            sO += Sred[tg][3][q][e];
        }
        float iv = sI + bias[ng];
        float fv = sF + bias[HDIM + ng];
        float gv = sG + bias[2 * HDIM + ng];
        float ov = sO + bias[3 * HDIM + ng];
        iv = 1.0f / (1.0f + __expf(-iv));
        fv = 1.0f / (1.0f + __expf(-fv));
        gv = tanhf(gv);
        ov = 1.0f / (1.0f + __expf(-ov));
        const int b = ib * 16 + m;
        const size_t off = (size_t)b * HDIM + ng;
        const float cn = fv * c[off] + iv * gv;
        c[off]     = cn;
        h_out[off] = (__bf16)(ov * tanhf(cn));
    }
}

// ---------------------------------------------------------------------------
extern "C" void kernel_launch(void* const* d_in, const int* in_sizes, int n_in,
                              void* d_out, int out_size, void* d_ws, size_t ws_size,
                              hipStream_t stream) {
    (void)in_sizes; (void)n_in; (void)out_size; (void)ws_size;
    const float* x    = (const float*)d_in[0];
    const float* Wih1 = (const float*)d_in[1];
    const float* Whh1 = (const float*)d_in[2];
    const float* bih1 = (const float*)d_in[3];
    const float* bhh1 = (const float*)d_in[4];
    const float* Wih2 = (const float*)d_in[5];
    const float* Whh2 = (const float*)d_in[6];
    const float* bih2 = (const float*)d_in[7];
    const float* bhh2 = (const float*)d_in[8];
    float* out = (float*)d_out;

    const int  BH  = BSZ * HDIM;                 // 65536
    const long XEL = (long)BSZ * TSTEPS * INDIM; // 8388608
    char* ws = (char*)d_ws;
    size_t off = 0;
    auto alloc = [&](size_t bytes) -> void* {
        void* p = ws + off;
        off = (off + bytes + 255) & ~(size_t)255;
        return p;
    };
    __bf16* Wcat1 = (__bf16*)alloc((size_t)4 * HDIM * 640  * 2);  // [2048][640]
    __bf16* Wcat2 = (__bf16*)alloc((size_t)4 * HDIM * 1024 * 2);  // [2048][1024]
    float*  bias1 = (float*) alloc((size_t)4 * HDIM * 4);
    float*  bias2 = (float*) alloc((size_t)4 * HDIM * 4);
    __bf16* xbf   = (__bf16*)alloc((size_t)XEL * 2);              // bf16 x
    __bf16* h1a   = (__bf16*)alloc((size_t)BH * 2);
    __bf16* h1b   = (__bf16*)alloc((size_t)BH * 2);
    __bf16* h2a   = (__bf16*)alloc((size_t)BH * 2);
    __bf16* h2b   = (__bf16*)alloc((size_t)BH * 2);
    float*  c1    = (float*) alloc((size_t)BH * 4);

    float* c2 = out + BH;   // c2 lives in d_out's second half (updated in place)

    // --- prep: bf16 x + weight packs, combined biases, zeroed state -------
    {
        long tot1 = (long)4 * HDIM * 640;
        long tot2 = (long)4 * HDIM * 1024;
        prep_weights_kernel<<<(int)((tot1 + 255) / 256), 256, 0, stream>>>(Wih1, Whh1, Wcat1, 640, 128, tot1);
        prep_weights_kernel<<<(int)((tot2 + 255) / 256), 256, 0, stream>>>(Wih2, Whh2, Wcat2, 1024, 512, tot2);
        prep_bias_kernel<<<(4 * HDIM + 255) / 256, 256, 0, stream>>>(bih1, bhh1, bias1, 4 * HDIM);
        prep_bias_kernel<<<(4 * HDIM + 255) / 256, 256, 0, stream>>>(bih2, bhh2, bias2, 4 * HDIM);
        f32_to_bf16_kernel<<<(int)((XEL + 255) / 256), 256, 0, stream>>>(x, xbf, XEL);
        // h1a,h1b,h2a,h2b contiguous bf16 (each 256B-multiple): one zero pass
        zero_bf16_kernel<<<(4 * BH + 255) / 256, 256, 0, stream>>>(h1a, 4 * BH);
        zero_f32_kernel<<<(BH + 255) / 256, 256, 0, stream>>>(c1, BH);
        zero_f32_kernel<<<(BH + 255) / 256, 256, 0, stream>>>(c2, BH);
    }

    __bf16* h1buf[2] = { h1a, h1b };
    __bf16* h2buf[2] = { h2a, h2b };

    // --- interleaved recurrence: L1 step t feeds L2 step t -----------------
    for (int t = 0; t < TSTEPS; ++t) {
        const __bf16* h1_in  = h1buf[t & 1];
        __bf16*       h1_out = h1buf[(t + 1) & 1];
        lstm_step_kernel<640, 128><<<128, 256, 0, stream>>>(
            xbf, (long)TSTEPS * INDIM, (long)t * INDIM,
            h1_in, h1_out, c1, Wcat1, bias1);

        const __bf16* h2_in  = h2buf[t & 1];
        __bf16*       h2_out = h2buf[(t + 1) & 1];
        lstm_step_kernel<1024, 512><<<128, 256, 0, stream>>>(
            h1_out, HDIM, 0,
            h2_in, h2_out, c2, Wcat2, bias2);
    }
    // T even -> final h2 is in h2buf[0]; widen to fp32 into d_out[0:BH].
    bf16_to_f32_kernel<<<(BH + 255) / 256, 256, 0, stream>>>(h2buf[0], out, BH);
}
